// FieldEvolution_85658827751545
// MI455X (gfx1250) — compile-verified, matrix operand order checked by reading, not run on hardware
//
#include <hip/hip_runtime.h>
#include <math.h>
#include <stdint.h>

typedef __bf16 bf16_t;
typedef bf16_t v16bf __attribute__((ext_vector_type(16)));
typedef float  v8f   __attribute__((ext_vector_type(8)));
typedef unsigned int u32x4 __attribute__((ext_vector_type(4)));

#define SEQ    2048
#define DIM    256
#define NBATCH 4
#define NTOT   (NBATCH * SEQ * DIM)
#define G_CONST 0.01f
#define DT      0.1f
#define EPSV    1e-8f

union Frag {
    v16bf v;
    u32x4 q[2];
    unsigned short u[16];
};

__device__ __forceinline__ unsigned short f2bf(float x) {
    union { float f; unsigned int i; } c; c.f = x;
    unsigned int u = c.i;
    unsigned int r = u + 0x7FFFu + ((u >> 16) & 1u);   // round-to-nearest-even
    return (unsigned short)(r >> 16);
}
__device__ __forceinline__ float bf2f(unsigned short h) {
    union { float f; unsigned int i; } c; c.i = ((unsigned int)h) << 16;
    return c.f;
}
// 16-byte-aligned load of one v16bf fragment: elements 0..7 from p[0..7],
// elements 8..15 from p[16..23].
__device__ __forceinline__ v16bf load_frag(const unsigned short* __restrict__ p) {
    Frag f;
    f.q[0] = *(const u32x4*)(p);
    f.q[1] = *(const u32x4*)(p + 16);
    return f.v;
}

// ---- prep: one wave per row. Split f into bf16 hi/lo planes, a transposed
// hi plane (for force-GEMM B fragments), and per-row sum of squares. ----
__global__ __launch_bounds__(256) void grav_prep_kernel(
    const float* __restrict__ f,
    unsigned short* __restrict__ fhi, unsigned short* __restrict__ flo,
    unsigned short* __restrict__ fThi, float* __restrict__ sq)
{
    const int lane = threadIdx.x & 31;
    const int row  = blockIdx.x * (blockDim.x >> 5) + (threadIdx.x >> 5); // 0..8191
    const int b    = row >> 11;
    const int s    = row & (SEQ - 1);

    const float* __restrict__ p = f + (size_t)row * DIM + lane * 8;
    unsigned short h[8], l[8];
    float ss = 0.f;
    #pragma unroll
    for (int e = 0; e < 8; ++e) {
        float x = p[e];
        unsigned short hh = f2bf(x);
        h[e] = hh;
        l[e] = f2bf(x - bf2f(hh));
        ss += x * x;
    }
    const size_t off = (size_t)row * DIM + lane * 8;
    #pragma unroll
    for (int e = 0; e < 8; ++e) { fhi[off + e] = h[e]; flo[off + e] = l[e]; }
    #pragma unroll
    for (int e = 0; e < 8; ++e) {
        fThi[((size_t)b * DIM + lane * 8 + e) * SEQ + s] = h[e];
    }
    ss += __shfl_xor(ss, 1, 32);
    ss += __shfl_xor(ss, 2, 32);
    ss += __shfl_xor(ss, 4, 32);
    ss += __shfl_xor(ss, 8, 32);
    ss += __shfl_xor(ss, 16, 32);
    if (lane == 0) sq[row] = ss;
}

// ---- hot kernel: one wave per 16-row i-tile, full D=256 output.
// out = base + coef * (G(f) @ f), optionally row-normalized.
// Gram computed transposed (A=f_j, B=f_i) so its C fragment (i in lane,
// j in regs) is directly the A fragment of G for the force GEMM. ----
__global__ __launch_bounds__(128, 1) void grav_force_kernel(
    const unsigned short* __restrict__ fhi, const unsigned short* __restrict__ flo,
    const unsigned short* __restrict__ fThi, const float* __restrict__ sq,
    const float* __restrict__ fbase, float* __restrict__ out,
    float coef, int do_norm)
{
    const int lane = threadIdx.x & 31;
    const int wave = blockIdx.x * (blockDim.x >> 5) + (threadIdx.x >> 5);
    const int b    = wave >> 7;          // 128 i-tiles per batch
    const int it   = wave & 127;
    const int i0   = it << 4;
    const int col  = lane & 15;
    const int koff = (lane & 16) ? 8 : 0;
    const size_t bS = (size_t)b * SEQ;

    // ---- preload f_i B fragments (hi/lo) ----
    v16bf Bi_hi[8], Bi_lo[8];
    {
        const unsigned short* rowi_hi = fhi + (bS + i0 + col) * DIM + koff;
        const unsigned short* rowi_lo = flo + (bS + i0 + col) * DIM + koff;
        #pragma unroll
        for (int kb = 0; kb < 8; ++kb) {
            Bi_hi[kb] = load_frag(rowi_hi + kb * 32);
            Bi_lo[kb] = load_frag(rowi_lo + kb * 32);
        }
    }
    const float sqi = sq[bS + i0 + col];

    const v8f vzero = {0.f, 0.f, 0.f, 0.f, 0.f, 0.f, 0.f, 0.f};
    v8f acc[16];
    #pragma unroll
    for (int dt = 0; dt < 16; ++dt) acc[dt] = vzero;

    // running pointers advanced by 32 j's per chunk
    const unsigned short* pj_hi = fhi + (bS + col) * DIM + koff;               // +jc*32*DIM
    const unsigned short* pj_lo = flo + (bS + col) * DIM + koff;
    const unsigned short* pT    = fThi + ((size_t)b * DIM + col) * SEQ + koff; // +jc*32
    const float* psq = sq + bS + col;                                          // +jc*32

    for (int jc = 0; jc < 64; ++jc) {
        // pull next chunk toward the near caches while this chunk computes
        __builtin_prefetch(pj_hi + 32 * DIM, 0, 3);
        __builtin_prefetch(pj_lo + 32 * DIM, 0, 3);
        __builtin_prefetch(pT + 32, 0, 3);

        Frag GA;                               // A fragment of G (M=i, K=32 j's)
        #pragma unroll
        for (int t = 0; t < 2; ++t) {
            const unsigned short* rj_hi = pj_hi + (size_t)t * 16 * DIM;
            const unsigned short* rj_lo = pj_lo + (size_t)t * 16 * DIM;
            // stage all fragments for this tile first (clustered b128 loads)...
            v16bf Ah[8], Al[8];
            #pragma unroll
            for (int kb = 0; kb < 8; ++kb) {
                Ah[kb] = load_frag(rj_hi + kb * 32);
                Al[kb] = load_frag(rj_lo + kb * 32);
            }
            // fence: keep the whole load group issued here (distinct regs ->
            // partial loadcnt waits; WMMA burst overlaps the load tail)
            __builtin_amdgcn_sched_barrier(0);
            // ...then burst the 24 gram WMMAs (split-bf16: hh + hl + lh)
            v8f c = vzero;
            #pragma unroll
            for (int kb = 0; kb < 8; ++kb) {
                c = __builtin_amdgcn_wmma_f32_16x16x32_bf16(false, Ah[kb], false, Bi_hi[kb], (short)0, c, false, false);
                c = __builtin_amdgcn_wmma_f32_16x16x32_bf16(false, Ah[kb], false, Bi_lo[kb], (short)0, c, false, false);
                c = __builtin_amdgcn_wmma_f32_16x16x32_bf16(false, Al[kb], false, Bi_hi[kb], (short)0, c, false, false);
            }
            const float sqj = psq[t * 16];
            const int jb = jc * 32 + t * 16;
            // c[r] = <f_j, f_i> for j = jb + r + koff, i = i0 + col
            #pragma unroll
            for (int r = 0; r < 8; ++r) {
                int jl = r + koff;
                int jg = jb + jl;
                float sqj_r = __shfl(sqj, jl, 32);
                float d2 = sqi + sqj_r - 2.0f * c[r];
                d2 = d2 > 0.f ? d2 : 0.f;
                float dist = sqrtf(d2) + EPSV;
                float g = (jg == (i0 + col)) ? 0.f : (G_CONST / dist);
                GA.u[t * 8 + r] = f2bf(g);
            }
        }
        // force accumulation: stage all 16 B fragments, then 16 independent
        // WMMAs (distinct accumulators, shared A) back-to-back.
        v16bf Bf[16];
        #pragma unroll
        for (int dt = 0; dt < 16; ++dt) Bf[dt] = load_frag(pT + (size_t)dt * 16 * SEQ);
        __builtin_amdgcn_sched_barrier(0);
        #pragma unroll
        for (int dt = 0; dt < 16; ++dt) {
            acc[dt] = __builtin_amdgcn_wmma_f32_16x16x32_bf16(false, GA.v, false, Bf[dt], (short)0, acc[dt], false, false);
        }
        pj_hi += 32 * DIM;
        pj_lo += 32 * DIM;
        pT    += 32;
        psq   += 32;
    }

    // ---- epilogue: out = base + coef*force ; optional row-normalize ----
    const float* __restrict__ baseb = fbase + bS * DIM;
    float* __restrict__ outb = out + bS * DIM;

    float rss[8];
    #pragma unroll
    for (int r = 0; r < 8; ++r) rss[r] = 0.f;

    #pragma unroll
    for (int dt = 0; dt < 16; ++dt) {
        #pragma unroll
        for (int r = 0; r < 8; ++r) {
            int m = r + koff;
            int d = dt * 16 + col;
            float v = baseb[(size_t)(i0 + m) * DIM + d] + coef * acc[dt][r];
            acc[dt][r] = v;
            rss[r] += v * v;
        }
    }
    if (do_norm) {
        #pragma unroll
        for (int r = 0; r < 8; ++r) {
            float t = rss[r];
            t += __shfl_xor(t, 1, 32);
            t += __shfl_xor(t, 2, 32);
            t += __shfl_xor(t, 4, 32);
            t += __shfl_xor(t, 8, 32);
            float inv = 1.0f / (sqrtf(t) + EPSV);
            #pragma unroll
            for (int dt = 0; dt < 16; ++dt) acc[dt][r] *= inv;
        }
    }
    #pragma unroll
    for (int dt = 0; dt < 16; ++dt) {
        #pragma unroll
        for (int r = 0; r < 8; ++r) {
            outb[(size_t)(i0 + r + koff) * DIM + dt * 16 + col] = acc[dt][r];
        }
    }
}

extern "C" void kernel_launch(void* const* d_in, const int* in_sizes, int n_in,
                              void* d_out, int out_size, void* d_ws, size_t ws_size,
                              hipStream_t stream) {
    (void)in_sizes; (void)n_in; (void)out_size; (void)ws_size;
    const float* fin = (const float*)d_in[0];
    float* out = (float*)d_out;

    // workspace layout (~20 MB): H | f_hi | f_lo | fT_hi | sq
    float* H = (float*)d_ws;                               // NTOT f32  (8 MB)
    unsigned short* fhi = (unsigned short*)(H + NTOT);     // NTOT bf16 (4 MB)
    unsigned short* flo = fhi + NTOT;                      // NTOT bf16 (4 MB)
    unsigned short* fT  = flo + NTOT;                      // NTOT bf16 (4 MB)
    float* sq = (float*)(fT + NTOT);                       // 8192 f32  (32 KB)

    const dim3 pgrid(1024), pblock(256);   // 8192 waves: one per row
    const dim3 fgrid(128),  fblock(128);   // 512 waves: one per 16-row i-tile

    // step 0 (must not mutate d_in): half-step -> H, full step -> out
    grav_prep_kernel<<<pgrid, pblock, 0, stream>>>(fin, fhi, flo, fT, sq);
    grav_force_kernel<<<fgrid, fblock, 0, stream>>>(fhi, flo, fT, sq, fin, H, 0.5f * DT, 0);
    grav_prep_kernel<<<pgrid, pblock, 0, stream>>>(H, fhi, flo, fT, sq);
    grav_force_kernel<<<fgrid, fblock, 0, stream>>>(fhi, flo, fT, sq, fin, out, DT, 1);
    // steps 1..9: in-place on out (each element read/written by its own lane)
    for (int s = 1; s < 10; ++s) {
        grav_prep_kernel<<<pgrid, pblock, 0, stream>>>(out, fhi, flo, fT, sq);
        grav_force_kernel<<<fgrid, fblock, 0, stream>>>(fhi, flo, fT, sq, out, H, 0.5f * DT, 0);
        grav_prep_kernel<<<pgrid, pblock, 0, stream>>>(H, fhi, flo, fT, sq);
        grav_force_kernel<<<fgrid, fblock, 0, stream>>>(fhi, flo, fT, sq, out, out, DT, 1);
    }
}
